// SynthesisNetwork_91268055040378
// MI455X (gfx1250) — compile-verified, matrix operand order checked by reading, not run on hardware
//
#include <hip/hip_runtime.h>
#include <math.h>

typedef __attribute__((ext_vector_type(16))) __bf16 v16bf;
typedef __attribute__((ext_vector_type(8)))  __bf16 v8bf;
typedef __attribute__((ext_vector_type(8)))  float  v8f;

#define NB 4
#define WDIM 512
#define GAIN 1.4142135623730951f

// ---------------------------------------------------------------- kernels

__global__ void k_tile_const(const float* __restrict__ c, float* __restrict__ out) {
  int idx = blockIdx.x * blockDim.x + threadIdx.x;
  if (idx >= NB * 4 * 4 * 512) return;
  out[idx] = c[idx & 8191];
}

__global__ void k_affine(const float* __restrict__ dlat, int dlatIdx,
                         const float* __restrict__ aw, const float* __restrict__ ab,
                         float* __restrict__ sout, int Cin) {
  int idx = blockIdx.x * blockDim.x + threadIdx.x;
  if (idx >= NB * Cin) return;
  int n = idx / Cin, c = idx % Cin;
  const float* d = dlat + (size_t)(n * 14 + dlatIdx) * WDIM;
  float acc = 0.f;
  for (int k = 0; k < WDIM; ++k) acc += d[k] * aw[k * Cin + c];
  sout[idx] = acc * 0.04419417382415922f + ab[c];   // 1/sqrt(512)
}

__global__ void k_wsq(const float* __restrict__ w, float* __restrict__ wsq,
                      int Cin, int Cout) {
  int idx = blockIdx.x * blockDim.x + threadIdx.x;
  if (idx >= Cin * Cout) return;
  int ci = idx / Cout, co = idx % Cout;
  float acc = 0.f;
  for (int t = 0; t < 9; ++t) {
    float v = w[(t * Cin + ci) * Cout + co];
    acc += v * v;
  }
  wsq[idx] = acc / (9.0f * (float)Cin);   // equalized-lr scale folded in (squared)
}

__global__ void k_demod(const float* __restrict__ s, const float* __restrict__ wsq,
                        float* __restrict__ dcoef, int Cin, int Cout) {
  int idx = blockIdx.x * blockDim.x + threadIdx.x;
  if (idx >= NB * Cout) return;
  int n = idx / Cout, co = idx % Cout;
  float acc = 1e-8f;
  for (int ci = 0; ci < Cin; ++ci) {
    float sv = s[n * Cin + ci];
    acc += sv * sv * wsq[ci * Cout + co];
  }
  dcoef[idx] = rsqrtf(acc);
}

// FIR upsample by 2 (4-tap separable, normalized 2D filter in `fir`, gain 4)
__global__ void k_upsample2x(const float* __restrict__ in, const float* __restrict__ fir,
                             float* __restrict__ out, int Hin, int C) {
  int Ho = Hin * 2;
  int total = NB * Ho * Ho * C;
  int idx = blockIdx.x * blockDim.x + threadIdx.x;
  if (idx >= total) return;
  int c = idx % C; int t = idx / C;
  int xo = t % Ho; t /= Ho;
  int yo = t % Ho; int n = t / Ho;
  int ry[2], iy[2], rx[2], ix[2];
  if (yo & 1) { iy[0] = 1; iy[1] = 3; ry[0] = (yo - 1) >> 1; ry[1] = (yo + 1) >> 1; }
  else        { iy[0] = 0; iy[1] = 2; ry[0] = (yo >> 1) - 1; ry[1] = yo >> 1; }
  if (xo & 1) { ix[0] = 1; ix[1] = 3; rx[0] = (xo - 1) >> 1; rx[1] = (xo + 1) >> 1; }
  else        { ix[0] = 0; ix[1] = 2; rx[0] = (xo >> 1) - 1; rx[1] = xo >> 1; }
  float acc = 0.f;
  for (int a = 0; a < 2; ++a) {
    if (ry[a] < 0 || ry[a] >= Hin) continue;
    for (int b2 = 0; b2 < 2; ++b2) {
      if (rx[b2] < 0 || rx[b2] >= Hin) continue;
      acc += fir[iy[a] * 4 + ix[b2]] * in[((n * Hin + ry[a]) * Hin + rx[b2]) * C + c];
    }
  }
  out[idx] = acc * 4.f;
}

// Swizzle fp32 weights -> bf16 B-fragments:
// layout [tap][kblock(32)][nblock(16)][lane(32)][elem(16)], matching the
// 16-bit WMMA operand VGPR mapping: k = kb*32 + (e>=8)*16 + (lane>=16)*8 + (e&7)
__global__ void k_weight_prep(const float* __restrict__ w, __bf16* __restrict__ wb,
                              int Cin, int Cout) {
  int total = 9 * (Cin >> 5) * (Cout >> 4) * 512;
  int idx = blockIdx.x * blockDim.x + threadIdx.x;
  if (idx >= total) return;
  int e = idx & 15;
  int l = (idx >> 4) & 31;
  int frag = idx >> 9;
  int nB = Cout >> 4, kB = Cin >> 5;
  int nb = frag % nB; int t2 = frag / nB;
  int kb = t2 % kB;  int t  = t2 / kB;
  int k  = kb * 32 + ((e >= 8) ? 16 : 0) + ((l >= 16) ? 8 : 0) + (e & 7);
  int co = nb * 16 + (l & 15);
  float v = w[(t * Cin + k) * Cout + co] * rsqrtf(9.f * (float)Cin);
  wb[idx] = (__bf16)v;
}

// Direct (VALU) modulated conv for tiny resolutions (4x4, 8x8)
__global__ void k_conv3x3_direct(const float* __restrict__ x, const float* __restrict__ w,
                                 const float* __restrict__ s, const float* __restrict__ dcoef,
                                 const float* __restrict__ noise, const float* __restrict__ nsp,
                                 const float* __restrict__ bias, float* __restrict__ out,
                                 int H, int Cin, int Cout) {
  int total = NB * H * H * Cout;
  int idx = blockIdx.x * blockDim.x + threadIdx.x;
  if (idx >= total) return;
  int co = idx % Cout; int t = idx / Cout;
  int xx = t % H; t /= H;
  int yy = t % H; int n = t / H;
  float wscale = rsqrtf(9.f * (float)Cin);
  const float* sp = s + n * Cin;
  float acc = 0.f;
  for (int ky = 0; ky < 3; ++ky) {
    int ys = yy + ky - 1;
    if (ys < 0 || ys >= H) continue;
    for (int kx = 0; kx < 3; ++kx) {
      int xs = xx + kx - 1;
      if (xs < 0 || xs >= H) continue;
      const float* xp = x + ((n * H + ys) * H + xs) * Cin;
      const float* wp = w + (ky * 3 + kx) * Cin * Cout + co;
      for (int ci = 0; ci < Cin; ++ci)
        acc += xp[ci] * sp[ci] * wp[ci * Cout];
    }
  }
  acc = acc * wscale * dcoef[n * Cout + co];
  acc += noise[(n * H + yy) * H + xx] * nsp[0] + bias[co];
  out[idx] = (acc > 0.f ? acc : 0.2f * acc) * GAIN;
}

// Fused implicit-GEMM 3x3 modulated conv via v_wmma_f32_16x16x32_bf16.
// WG = 256 threads = 8 waves; tile = 2 rows x 16 px x 128 couts.
// Each wave: two M tiles (rows y0, y0+1) sharing one B fragment -> 2 WMMAs per
// B load into independent accumulators; B pointer advances linearly.
__global__ __launch_bounds__(256) void k_conv3x3_wmma(
    const float* __restrict__ x, const __bf16* __restrict__ wb,
    const float* __restrict__ s, const float* __restrict__ dcoef,
    const float* __restrict__ noise, const float* __restrict__ nsp,
    const float* __restrict__ bias, float* __restrict__ out,
    int H, int cinShift, int Cout) {
  const int Cin = 1 << cinShift;
  extern __shared__ char smem_raw[];
  __bf16* lds = (__bf16*)smem_raw;          // [4 rows][18 x][Cin] bf16

  unsigned by = blockIdx.y;
  int n  = by / (unsigned)(H >> 1);
  int y0 = (by % (unsigned)(H >> 1)) * 2;
  int x0 = blockIdx.x * 16;
  unsigned tid = threadIdx.x;

  // Stage style-scaled input patch (rows y0-1 .. y0+2) into LDS as bf16.
  const float* sn = s + (n << cinShift);
  int total = (4 * 18) << cinShift;
  for (int idx = tid; idx < total; idx += 256) {
    int ci = idx & (Cin - 1);
    int p  = idx >> cinShift;
    int xi = p % 18;  int r = p / 18;
    int yy = y0 + r - 1; int xx = x0 + xi - 1;
    float v = 0.f;
    if (yy >= 0 && yy < H && xx >= 0 && xx < H)
      v = x[(((size_t)(n * H + yy) * H + xx) << cinShift) + ci] * sn[ci];
    lds[idx] = (__bf16)v;
  }
  __syncthreads();

  unsigned wave = tid >> 5, lane = tid & 31;
  unsigned m = lane & 15, hi = lane >> 4;    // A: lane -> pixel, hi -> K half
  int coutBase = blockIdx.z * 128 + wave * 16;
  int kB = Cin >> 5, nB = Cout >> 4;

  v8f acc0 = {}, acc1 = {};
  // Linear B pointer: fragment index (t*kB+kb) increments by 1 per k-step.
  const __bf16* bp = wb + (size_t)(coutBase >> 4) * 512 + (size_t)lane * 16;
  const size_t bstride = (size_t)nB * 512;

  for (int t = 0; t < 9; ++t) {
    int ky = t / 3, kx = t - ky * 3;
    const __bf16* arow0 = lds + (((unsigned)(ky * 18 + kx) + m) << cinShift);
    const __bf16* arow1 = arow0 + (18 << cinShift);
    for (int kb = 0; kb < kB; ++kb) {
      int k0 = kb * 32 + hi * 8;
      v8bf a0 = *(const v8bf*)(arow0 + k0);        // K = k0..k0+7
      v8bf a1 = *(const v8bf*)(arow0 + k0 + 16);   // K = k0+16..k0+23
      v8bf c0 = *(const v8bf*)(arow1 + k0);
      v8bf c1 = *(const v8bf*)(arow1 + k0 + 16);
      v16bf A0, A1;
#pragma unroll
      for (int i = 0; i < 8; ++i) {
        A0[i] = a0[i]; A0[i + 8] = a1[i];
        A1[i] = c0[i]; A1[i + 8] = c1[i];
      }
      v16bf B = *(const v16bf*)bp;                 // 32B contiguous per lane
      acc0 = __builtin_amdgcn_wmma_f32_16x16x32_bf16(
                 false, A0, false, B, (short)0, acc0, false, false);
      acc1 = __builtin_amdgcn_wmma_f32_16x16x32_bf16(
                 false, A1, false, B, (short)0, acc1, false, false);
      bp += bstride;
    }
  }

  // Epilogue: demod, noise, bias, leaky-relu * sqrt(2), two rows.
  int co = coutBase + m;                 // C/D: lane -> N column
  float dc  = dcoef[n * Cout + co];
  float bv  = bias[co];
  float nsv = nsp[0];
#pragma unroll
  for (int r = 0; r < 8; ++r) {
    int mm = r + hi * 8;                 // C/D: vgpr -> M row (+8 for upper lanes)
    int xx = x0 + mm;
    float v0 = acc0[r] * dc + noise[(n * H + y0) * H + xx] * nsv + bv;
    v0 = (v0 > 0.f ? v0 : 0.2f * v0) * GAIN;
    out[((size_t)((n * H + y0) * H + xx) * Cout) + co] = v0;
    float v1 = acc1[r] * dc + noise[(n * H + y0 + 1) * H + xx] * nsv + bv;
    v1 = (v1 > 0.f ? v1 : 0.2f * v1) * GAIN;
    out[((size_t)((n * H + y0 + 1) * H + xx) * Cout) + co] = v1;
  }
}

__global__ void k_torgb(const float* __restrict__ x, const float* __restrict__ s,
                        const float* __restrict__ w, const float* __restrict__ b,
                        const float* __restrict__ yup, float* __restrict__ out,
                        int H, int Cin) {
  int total = NB * H * H;
  int idx = blockIdx.x * blockDim.x + threadIdx.x;
  if (idx >= total) return;
  int n = idx / (H * H);
  const float* xp = x + (size_t)idx * Cin;
  const float* sp = s + n * Cin;
  float ws = rsqrtf((float)Cin);
  float a0 = 0.f, a1 = 0.f, a2 = 0.f;
  for (int ci = 0; ci < Cin; ++ci) {
    float xv = xp[ci] * sp[ci];
    a0 += xv * w[ci * 3 + 0];
    a1 += xv * w[ci * 3 + 1];
    a2 += xv * w[ci * 3 + 2];
  }
  float y0 = a0 * ws + b[0], y1 = a1 * ws + b[1], y2 = a2 * ws + b[2];
  if (yup) { y0 += yup[idx * 3 + 0]; y1 += yup[idx * 3 + 1]; y2 += yup[idx * 3 + 2]; }
  out[idx * 3 + 0] = y0; out[idx * 3 + 1] = y1; out[idx * 3 + 2] = y2;
}

// ---------------------------------------------------------------- host

extern "C" void kernel_launch(void* const* d_in, const int* in_sizes, int n_in,
                              void* d_out, int out_size, void* d_ws, size_t ws_size,
                              hipStream_t stream) {
  (void)in_sizes; (void)n_in; (void)out_size; (void)ws_size;
  const float* dlat = (const float*)d_in[0];
  const float* fir  = (const float*)d_in[1];
  const float* cst  = (const float*)d_in[2];
  auto F = [&](int i) { return (const float*)d_in[i]; };

  char* ws = (char*)d_ws;                          // bump-allocated scratch
  float*  ACT_A = (float*)(ws + (size_t)0);          // 33.5M f32
  float*  ACT_B = (float*)(ws + (size_t)134217728);  // 33.5M f32
  float*  UPB   = (float*)(ws + (size_t)268435456);  // 67.1M f32 (upsampled x)
  float*  YB_A  = (float*)(ws + (size_t)536870912);  // rgb accum
  float*  YB_B  = (float*)(ws + (size_t)540016640);  // rgb upsample tmp
  __bf16* WB    = (__bf16*)(ws + (size_t)543162368); // swizzled bf16 weights
  float*  WSQ   = (float*)(ws + (size_t)547880960);
  float*  SB    = (float*)(ws + (size_t)548929536);  // styles [4, Cin]
  float*  DB    = (float*)(ws + (size_t)548937728);  // demod  [4, Cout]

  static const int outCh[9] = {0, 0, 512, 512, 512, 512, 512, 256, 128};
  auto cdiv = [](int a, int b) { return (a + b - 1) / b; };

  auto run_layer = [&](int res, int li01, const float* xin, float* xout,
                       int Hin, bool up) {
    int H = 1 << res;
    int Cin  = (li01 == 0) ? outCh[res - 1] : outCh[res];
    int Cout = outCh[res];
    int lin  = (res == 2) ? 0 : 1 + (res - 3) * 2 + li01;   // alphabetical order
    int base = 3 + lin * 6;                                  // ab,aw,b,noise,ns,w
    const float* ab  = F(base + 0); const float* aw  = F(base + 1);
    const float* bb  = F(base + 2); const float* nz  = F(base + 3);
    const float* nsp = F(base + 4); const float* wPtr = F(base + 5);
    int dIdx = (res == 2) ? 0 : (2 * res - 5 + li01);

    k_affine<<<cdiv(NB * Cin, 256), 256, 0, stream>>>(dlat, dIdx, aw, ab, SB, Cin);
    k_wsq   <<<cdiv(Cin * Cout, 256), 256, 0, stream>>>(wPtr, WSQ, Cin, Cout);
    k_demod <<<cdiv(NB * Cout, 256), 256, 0, stream>>>(SB, WSQ, DB, Cin, Cout);

    const float* cin_x = xin;
    if (up) {
      int Ho = Hin * 2;
      k_upsample2x<<<cdiv(NB * Ho * Ho * Cin, 256), 256, 0, stream>>>(
          xin, fir, UPB, Hin, Cin);
      cin_x = UPB;
    }
    if (H >= 16) {
      int total = 9 * (Cin / 32) * (Cout / 16) * 512;
      k_weight_prep<<<cdiv(total, 256), 256, 0, stream>>>(wPtr, WB, Cin, Cout);
      int cinShift = __builtin_ctz((unsigned)Cin);
      dim3 grid(H / 16, NB * H / 2, Cout / 128);
      size_t shmem = (size_t)4 * 18 * Cin * sizeof(__bf16);
      k_conv3x3_wmma<<<grid, 256, shmem, stream>>>(
          cin_x, WB, SB, DB, nz, nsp, bb, xout, H, cinShift, Cout);
    } else {
      k_conv3x3_direct<<<cdiv(NB * H * H * Cout, 256), 256, 0, stream>>>(
          cin_x, wPtr, SB, DB, nz, nsp, bb, xout, H, Cin, Cout);
    }
  };

  auto run_torgb = [&](int res, const float* xin, const float* yup, float* yout) {
    int Cin = outCh[res]; int H = 1 << res;
    int base = 81 + (res - 2) * 4;                           // ab,aw,b,w
    const float* ab = F(base + 0); const float* aw = F(base + 1);
    const float* bb = F(base + 2); const float* wPtr = F(base + 3);
    int dIdx = 2 * res - 3;
    k_affine<<<cdiv(NB * Cin, 256), 256, 0, stream>>>(dlat, dIdx, aw, ab, SB, Cin);
    k_torgb <<<cdiv(NB * H * H, 256), 256, 0, stream>>>(xin, SB, wPtr, bb, yup, yout, H, Cin);
  };

  // res 2: const -> layer -> torgb
  k_tile_const<<<cdiv(NB * 8192, 256), 256, 0, stream>>>(cst, ACT_A);
  run_layer(2, 0, ACT_A, ACT_B, 4, false);
  float* cur = ACT_B; float* oth = ACT_A;
  run_torgb(2, cur, nullptr, YB_A);

  for (int res = 3; res <= 8; ++res) {
    int Hin = 1 << (res - 1);
    run_layer(res, 0, cur, oth, Hin, true);
    { float* t = cur; cur = oth; oth = t; }
    run_layer(res, 1, cur, oth, 0, false);
    { float* t = cur; cur = oth; oth = t; }
    int Ho = 1 << res;
    k_upsample2x<<<cdiv(NB * Ho * Ho * 3, 256), 256, 0, stream>>>(
        YB_A, fir, YB_B, Hin, 3);
    run_torgb(res, cur, YB_B, (res == 8) ? (float*)d_out : YB_A);
  }
}
